// LearnableMetricSimilarity_26061861552520
// MI455X (gfx1250) — matrix operand order repaired and checked by reading, not compiled, not run on hardware
//
#include <hip/hip_runtime.h>
#include <hip/hip_bf16.h>
#include <cstdint>
#include <cstddef>

// ---------------------------------------------------------------------------
// LearnableMetricSimilarity for MI455X (gfx1250, wave32, WMMA)
//   out[b,i,j] = exp(clamp(-(q1[b,i]+q2[b,j]-2*cross[b,i,j]), -10, 10))
//   f1 = x1@W^T+b ; f1M = x1@(W^T Msym)+b Msym ; cross = f1M @ f2^T
// All GEMMs run as split-bf16 (hi+lo) v_wmma_f32_16x16x32_bf16 with fp32 acc.
// ---------------------------------------------------------------------------

typedef __bf16 bf16_t;
typedef __bf16 v16bf __attribute__((ext_vector_type(16)));
typedef __bf16 v8bf  __attribute__((ext_vector_type(8)));
typedef float  v8f   __attribute__((ext_vector_type(8)));

#define BF16_WMMA(A, B, C)                                                    \
  __builtin_amdgcn_wmma_f32_16x16x32_bf16(false, (A), false, (B), (short)0,   \
                                          (C), false, false)

constexpr int Dk   = 256;    // feature dim
constexpr int Nrow = 2048;   // N1 == N2
constexpr int Bb   = 8;      // batch
constexpr int Rtot = Bb * Nrow;  // 16384 rows per feature set

// ---------------------------------------------------------------------------
// Fragment loaders (row-major sources, row stride = Dk elements).
// A-layout (16x32 bf16): lane L -> M=L&15, K = kb + (L>>4)*16 + 0..15 contiguous.
// B-layout (32x16 bf16): lane L -> N=L&15, K = kb + (L>>4)*16 + 0..15 contiguous.
// ---------------------------------------------------------------------------
__device__ __forceinline__ v16bf load_a_frag(const bf16_t* __restrict__ tile_base,
                                             int lane, int kb) {
  int row = lane & 15;
  int kk  = kb + ((lane >> 4) << 4);
  const bf16_t* p = tile_base + (size_t)row * Dk + kk;
  return *(const v16bf*)(p);
}

__device__ __forceinline__ v16bf load_b_frag(const bf16_t* __restrict__ tile_base,
                                             int lane, int kb) {
  int row = lane & 15;
  int kk  = kb + ((lane >> 4) << 4);
  return *(const v16bf*)(tile_base + (size_t)row * Dk + kk);
}

// ---------------------------------------------------------------------------
// Kernel 0: Q = Msym @ W (row-major: Q[n,k]), bM[n] = Msym row n . b,
// plus bf16 hi/lo split of W and Q. 256 blocks x 256 threads, thread=(n,k).
// ---------------------------------------------------------------------------
__global__ void __launch_bounds__(256)
k_prep(const float* __restrict__ W, const float* __restrict__ bvec,
       const float* __restrict__ M,
       bf16_t* __restrict__ Whi, bf16_t* __restrict__ Wlo,
       bf16_t* __restrict__ Qhi, bf16_t* __restrict__ Qlo,
       float* __restrict__ bM) {
  int n = blockIdx.x;
  int k = threadIdx.x;
  float acc = 0.f;
  for (int j = 0; j < Dk; ++j) {
    float ms = 0.5f * (M[n * Dk + j] + M[j * Dk + n]);
    acc = fmaf(ms, W[j * Dk + k], acc);
  }
  bf16_t qh = (bf16_t)acc;
  Qhi[n * Dk + k] = qh;
  Qlo[n * Dk + k] = (bf16_t)(acc - (float)qh);

  float w = W[n * Dk + k];
  bf16_t wh = (bf16_t)w;
  Whi[n * Dk + k] = wh;
  Wlo[n * Dk + k] = (bf16_t)(w - (float)wh);

  if (k == 0) {
    float a2 = 0.f;
    for (int j = 0; j < Dk; ++j) {
      float ms = 0.5f * (M[n * Dk + j] + M[j * Dk + n]);
      a2 = fmaf(ms, bvec[j], a2);
    }
    bM[n] = a2;
  }
}

// ---------------------------------------------------------------------------
// Kernel 1: split fp32 features into bf16 hi/lo.
// ---------------------------------------------------------------------------
__global__ void __launch_bounds__(256)
k_split(const float* __restrict__ x, bf16_t* __restrict__ xhi,
        bf16_t* __restrict__ xlo, int total) {
  int i = blockIdx.x * 256 + threadIdx.x;
  if (i >= total) return;
  float v  = x[i];
  bf16_t h = (bf16_t)v;
  xhi[i] = h;
  xlo[i] = (bf16_t)(v - (float)h);
}

// ---------------------------------------------------------------------------
// Kernel 2 (stage 1): per-wave 16-row stripe. For each of 16 N-tiles:
//   f  = x @ W^T + b      (dual-B GEMM sharing A fragments)
//   g  = x @ Q^T + bM     (= f @ Msym)
// Accumulates q[m] = sum_n f*g, stores (store_g ? g : f) as bf16 hi/lo.
// 3 WMMAs per (tile, k-step) per output matrix (split-bf16).
// ---------------------------------------------------------------------------
__global__ void __launch_bounds__(128)
k_stage1(const bf16_t* __restrict__ xhi, const bf16_t* __restrict__ xlo,
         const bf16_t* __restrict__ Whi, const bf16_t* __restrict__ Wlo,
         const bf16_t* __restrict__ Qhi, const bf16_t* __restrict__ Qlo,
         const float* __restrict__ bvec, const float* __restrict__ bM,
         bf16_t* __restrict__ out_hi, bf16_t* __restrict__ out_lo,
         float* __restrict__ q, int store_g) {
  const int wave = threadIdx.x >> 5;
  const int lane = threadIdx.x & 31;
  const int half = lane >> 4;
  const int col  = lane & 15;
  const int m0   = (blockIdx.x * 4 + wave) << 4;

  const bf16_t* arow_hi = xhi + (size_t)m0 * Dk;
  const bf16_t* arow_lo = xlo + (size_t)m0 * Dk;

  float qacc[8] = {0.f, 0.f, 0.f, 0.f, 0.f, 0.f, 0.f, 0.f};

  for (int nt = 0; nt < 16; ++nt) {
    v8f accf = {};
    v8f accg = {};
    const bf16_t* wbh = Whi + (size_t)(nt * 16) * Dk;
    const bf16_t* wbl = Wlo + (size_t)(nt * 16) * Dk;
    const bf16_t* qbh = Qhi + (size_t)(nt * 16) * Dk;
    const bf16_t* qbl = Qlo + (size_t)(nt * 16) * Dk;
#pragma unroll
    for (int ks = 0; ks < 8; ++ks) {
      const int kb = ks * 32;
      v16bf ah  = load_a_frag(arow_hi, lane, kb);
      v16bf al  = load_a_frag(arow_lo, lane, kb);
      v16bf bwh = load_b_frag(wbh, lane, kb);
      v16bf bwl = load_b_frag(wbl, lane, kb);
      v16bf bqh = load_b_frag(qbh, lane, kb);
      v16bf bql = load_b_frag(qbl, lane, kb);
      accf = BF16_WMMA(ah, bwh, accf);
      accf = BF16_WMMA(ah, bwl, accf);
      accf = BF16_WMMA(al, bwh, accf);
      accg = BF16_WMMA(ah, bqh, accg);
      accg = BF16_WMMA(ah, bql, accg);
      accg = BF16_WMMA(al, bqh, accg);
    }
    const int n = nt * 16 + col;
    const float bb = bvec[n];
    const float bm = bM[n];
#pragma unroll
    for (int r = 0; r < 8; ++r) {
      float fv = accf[r] + bb;
      float gv = accg[r] + bm;
      qacc[r]  = fmaf(fv, gv, qacc[r]);
      float sv = store_g ? gv : fv;
      bf16_t sh = (bf16_t)sv;
      bf16_t sl = (bf16_t)(sv - (float)sh);
      size_t idx = (size_t)(m0 + half * 8 + r) * Dk + n;
      out_hi[idx] = sh;
      out_lo[idx] = sl;
    }
  }
  // Row-sum reduction across each 16-lane half (xor masks < 16 stay in-half).
#pragma unroll
  for (int r = 0; r < 8; ++r) {
    float v = qacc[r];
    v += __shfl_xor(v, 1, 32);
    v += __shfl_xor(v, 2, 32);
    v += __shfl_xor(v, 4, 32);
    v += __shfl_xor(v, 8, 32);
    if (col == 0) q[m0 + half * 8 + r] = v;
  }
}

// ---------------------------------------------------------------------------
// Kernel 3 (stage 2): cross GEMM + epilogue.
// Each wave owns a 32x32 output macro-tile of one batch:
//   acc[i][j] (fp32) = f1M_tile(i) @ f2_tile(j)^T   via 3 split-bf16 WMMAs
//   out = exp(clamp(-(q1+q2-2*acc), -10, 10))
// 8 waves / block, 4096 blocks -> 8 * 64 * 64 macro-tiles.
// ---------------------------------------------------------------------------
__global__ void __launch_bounds__(256)
k_stage2(const bf16_t* __restrict__ g_hi, const bf16_t* __restrict__ g_lo,
         const bf16_t* __restrict__ f_hi, const bf16_t* __restrict__ f_lo,
         const float* __restrict__ q1, const float* __restrict__ q2,
         float* __restrict__ out) {
  const int wave = threadIdx.x >> 5;
  const int lane = threadIdx.x & 31;
  const int half = lane >> 4;
  const int col  = lane & 15;

  const int flat = blockIdx.x * 8 + wave;   // 0..32767
  const int bat  = flat >> 12;              // 64*64 macro-tiles per batch
  const int rem  = flat & 4095;
  const int m0   = (rem >> 6) << 5;
  const int n0   = (rem & 63) << 5;

  const size_t arow = (size_t)(bat * Nrow + m0) * Dk;
  const size_t brow = (size_t)(bat * Nrow + n0) * Dk;
  const bf16_t* a0h = g_hi + arow;
  const bf16_t* a1h = a0h + 16 * Dk;
  const bf16_t* a0l = g_lo + arow;
  const bf16_t* a1l = a0l + 16 * Dk;
  const bf16_t* b0h = f_hi + brow;
  const bf16_t* b1h = b0h + 16 * Dk;
  const bf16_t* b0l = f_lo + brow;
  const bf16_t* b1l = b0l + 16 * Dk;

  v8f acc00 = {}, acc01 = {}, acc10 = {}, acc11 = {};

#pragma unroll
  for (int ks = 0; ks < 8; ++ks) {
    const int kb = ks * 32;
    if (ks < 7) {  // prefetch next K-step (global_prefetch_b8)
      const int kn = kb + 32;
      __builtin_prefetch(a0h + (size_t)(lane & 15) * Dk + kn, 0, 3);
      __builtin_prefetch(a1h + (size_t)(lane & 15) * Dk + kn, 0, 3);
      __builtin_prefetch(b0h + (size_t)(lane & 15) * Dk + kn, 0, 3);
      __builtin_prefetch(b1h + (size_t)(lane & 15) * Dk + kn, 0, 3);
    }
    v16bf ah0 = load_a_frag(a0h, lane, kb);
    v16bf ah1 = load_a_frag(a1h, lane, kb);
    v16bf al0 = load_a_frag(a0l, lane, kb);
    v16bf al1 = load_a_frag(a1l, lane, kb);
    v16bf bh0 = load_b_frag(b0h, lane, kb);
    v16bf bh1 = load_b_frag(b1h, lane, kb);
    v16bf bl0 = load_b_frag(b0l, lane, kb);
    v16bf bl1 = load_b_frag(b1l, lane, kb);

    acc00 = BF16_WMMA(ah0, bh0, acc00);
    acc00 = BF16_WMMA(ah0, bl0, acc00);
    acc00 = BF16_WMMA(al0, bh0, acc00);

    acc01 = BF16_WMMA(ah0, bh1, acc01);
    acc01 = BF16_WMMA(ah0, bl1, acc01);
    acc01 = BF16_WMMA(al0, bh1, acc01);

    acc10 = BF16_WMMA(ah1, bh0, acc10);
    acc10 = BF16_WMMA(ah1, bl0, acc10);
    acc10 = BF16_WMMA(al1, bh0, acc10);

    acc11 = BF16_WMMA(ah1, bh1, acc11);
    acc11 = BF16_WMMA(ah1, bl1, acc11);
    acc11 = BF16_WMMA(al1, bh1, acc11);
  }

  const float* q1p = q1 + bat * Nrow;
  const float* q2p = q2 + bat * Nrow;
  const size_t obase = (size_t)bat << 22;  // bat * 2048 * 2048
  const float LOG2E = 1.4426950408889634f;

#pragma unroll
  for (int i = 0; i < 2; ++i) {
    const v8f* accs = (i == 0) ? &acc00 : &acc10;  // [j stride via pointer]
#pragma unroll
    for (int j = 0; j < 2; ++j) {
      v8f a = (i == 0) ? (j == 0 ? acc00 : acc01) : (j == 0 ? acc10 : acc11);
      (void)accs;
      const int n  = n0 + j * 16 + col;
      const float q2v = q2p[n];
#pragma unroll
      for (int r = 0; r < 8; ++r) {
        const int m = m0 + i * 16 + half * 8 + r;
        float d = q1p[m] + q2v - 2.0f * a[r];
        float s = fminf(fmaxf(-d, -10.0f), 10.0f);
        out[obase + (size_t)m * Nrow + n] = exp2f(s * LOG2E);
      }
    }
  }
}

// ---------------------------------------------------------------------------
// Host-side launcher. All launches on `stream`; workspace carved from d_ws.
// ---------------------------------------------------------------------------
extern "C" void kernel_launch(void* const* d_in, const int* in_sizes, int n_in,
                              void* d_out, int out_size, void* d_ws,
                              size_t ws_size, hipStream_t stream) {
  const float* x1 = (const float*)d_in[0];  // (8,2048,256)
  const float* x2 = (const float*)d_in[1];  // (8,2048,256)
  const float* W  = (const float*)d_in[2];  // (256,256)
  const float* bv = (const float*)d_in[3];  // (256,)
  const float* M  = (const float*)d_in[4];  // (256,256)
  float* out = (float*)d_out;               // (8,2048,2048)

  char* ws = (char*)d_ws;
  size_t off = 0;
  auto carve = [&](size_t bytes) -> void* {
    void* p = ws + off;
    off = (off + bytes + 255) & ~(size_t)255;
    return p;
  };

  const size_t mat_bf  = (size_t)Dk * Dk * sizeof(bf16_t);        // 128 KB
  const size_t feat_bf = (size_t)Rtot * Dk * sizeof(bf16_t);      // 8 MB

  bf16_t* Whi  = (bf16_t*)carve(mat_bf);
  bf16_t* Wlo  = (bf16_t*)carve(mat_bf);
  bf16_t* Qhi  = (bf16_t*)carve(mat_bf);
  bf16_t* Qlo  = (bf16_t*)carve(mat_bf);
  float*  bM   = (float*)carve(Dk * sizeof(float));
  float*  q1   = (float*)carve(Rtot * sizeof(float));
  float*  q2   = (float*)carve(Rtot * sizeof(float));
  bf16_t* xhi  = (bf16_t*)carve(feat_bf);   // reused for x1 then x2
  bf16_t* xlo  = (bf16_t*)carve(feat_bf);
  bf16_t* g1hi = (bf16_t*)carve(feat_bf);   // f1M = f1 @ Msym
  bf16_t* g1lo = (bf16_t*)carve(feat_bf);
  bf16_t* f2hi = (bf16_t*)carve(feat_bf);   // f2
  bf16_t* f2lo = (bf16_t*)carve(feat_bf);

  const int feat_elems = Rtot * Dk;                 // 4,194,304
  const int split_blocks = feat_elems / 256;        // 16384

  // Precompute Q = Msym@W, bM, and W/Q hi-lo splits.
  k_prep<<<dim3(256), dim3(256), 0, stream>>>(W, bv, M, Whi, Wlo, Qhi, Qlo, bM);

  // Set 1: split x1, then f1M (stored) + q1.  (stream order serializes reuse)
  k_split<<<dim3(split_blocks), dim3(256), 0, stream>>>(x1, xhi, xlo, feat_elems);
  k_stage1<<<dim3(256), dim3(128), 0, stream>>>(xhi, xlo, Whi, Wlo, Qhi, Qlo,
                                                bv, bM, g1hi, g1lo, q1, 1);

  // Set 2: split x2 (reusing buffers), then f2 (stored) + q2.
  k_split<<<dim3(split_blocks), dim3(256), 0, stream>>>(x2, xhi, xlo, feat_elems);
  k_stage1<<<dim3(256), dim3(128), 0, stream>>>(xhi, xlo, Whi, Wlo, Qhi, Qlo,
                                                bv, bM, f2hi, f2lo, q2, 0);

  // Stage 2: cross GEMM + fused exp(clamp(-dist)) epilogue.
  k_stage2<<<dim3(4096), dim3(256), 0, stream>>>(g1hi, g1lo, f2hi, f2lo, q1, q2,
                                                 out);
}
